// ReverseLSTMLayer_68066641707650
// MI455X (gfx1250) — matrix operand order repaired and checked by reading, compiled, not run on hardware
//
#include <hip/hip_runtime.h>
#include <hip/hip_bf16.h>

// ---------------------------------------------------------------------------
// Time-aware reverse LSTM for MI455X (gfx1250, wave32, WMMA bf16).
//   T=512, B=64, D=256, H=512.  Gates G = 4H = 2048.  Kx = D+1 = 257 -> pad 288.
//
// Persistent-kernel design:
//   * ONE launch runs all 512 recurrent steps; steps separated by a
//     device-wide monotonic atomic barrier (32 blocks -> trivially resident).
//   * Block b owns hidden-column tile j0=16*b across ALL FOUR gate blocks;
//     wave w owns batch-row tile 16*w.  Epilogue is fully lane-local.
//   * W_hh slab (4 gates x 16 k-iters x 1KB = 64 KB) staged to LDS ONCE in
//     WMMA-fragment layout; W_hh is read from L2 exactly once per run.
//   * x_t tile (36 KB/block) is DOUBLE-BUFFERED in LDS via CDNA5 async
//     global->LDS copies (ASYNCcnt): next step's tile is issued BEFORE the
//     inter-step barrier so the copy overlaps barrier latency.  Falls back
//     to direct global loads + global_prefetch_b8 if the builtin is absent.
//   * bf16 WMMA (v_wmma_f32_16x16x32_bf16) with f32 accumulation.
// ---------------------------------------------------------------------------

typedef __bf16 bf16_t;
typedef __attribute__((ext_vector_type(16))) __bf16 v16bf;
typedef __attribute__((ext_vector_type(8)))  float  v8f;

static constexpr int Tn = 512;
static constexpr int Bn = 64;
static constexpr int Dn = 256;
static constexpr int Hn = 512;
static constexpr int Gn = 4 * Hn;      // 2048
static constexpr int DI = Dn + 1;      // 257
static constexpr int KP = 288;         // DI padded to multiple of 32
static constexpr int KPL = KP + 8;     // LDS x row pitch (elems): bank-spread

static constexpr int NBLK  = 32;       // = Hn/16 j-tiles
static constexpr int NWAVE = 4;        // = Bn/16 m-tiles
static constexpr int BLKTHREADS = NWAVE * 32;

// Async global->LDS path available?
#if defined(__has_builtin)
#if __has_builtin(__builtin_amdgcn_global_load_async_to_lds_b128) && \
    __has_builtin(__builtin_amdgcn_s_wait_asynccnt)
#define USE_ASYNC_X 1
#endif
#endif
#ifndef USE_ASYNC_X
#define USE_ASYNC_X 0
#endif

#if USE_ASYNC_X
// Builtin signature (leaked by clang diagnostic): param 1 is a GLOBAL-AS
// pointer to 'int __attribute__((vector_size(16)))' (non-const).
typedef int v4i __attribute__((vector_size(16)));
typedef __attribute__((address_space(1))) v4i* gas_v4i_p;   // global int4*
typedef __attribute__((address_space(3))) v4i* las_v4i_p;   // LDS int4*
#endif

// --------------------------- fragment loaders ------------------------------
// A fragment: 16(M) x 32(K) bf16 row-major.  Lanes 0-15 = rows holding
// K{0..7,16..23}; lanes 16-31 hold K{8..15,24..31}.
__device__ __forceinline__ v16bf load_a_frag(const bf16_t* __restrict__ base,
                                             int lda, int m0, int k0, int lane) {
    const int row  = m0 + (lane & 15);
    const int koff = (lane >> 4) * 8;
    const bf16_t* p = base + (size_t)row * lda + k0 + koff;
    union { v16bf v; struct { uint4 lo; uint4 hi; } q; } u;
    u.q.lo = *reinterpret_cast<const uint4*>(p);        // K +0..7   (16 B)
    u.q.hi = *reinterpret_cast<const uint4*>(p + 16);   // K +16..23 (16 B)
    return u.v;
}

// Same A fragment sourced from an LDS tile with row pitch KPL elems.
__device__ __forceinline__ v16bf load_a_frag_lds(const unsigned char* base,
                                                 int m0, int k0, int lane) {
    const int row  = m0 + (lane & 15);
    const int koff = (lane >> 4) * 8;
    const unsigned char* p = base + row * (KPL * 2) + (k0 + koff) * 2;
    union { v16bf v; struct { uint4 lo; uint4 hi; } q; } u;
    u.q.lo = *reinterpret_cast<const uint4*>(p);
    u.q.hi = *reinterpret_cast<const uint4*>(p + 32);   // +16 elems
    return u.v;
}

// B fragment: 32(K) x 16(N) bf16 with B[k][n] = W[n0+n][k0+k] (row-major W)
// -> computes A @ W^T.  Lane n (0-15): K{0..7,16..23}; lane n+16: K{8..15,24..31}.
__device__ __forceinline__ v16bf load_b_frag(const bf16_t* __restrict__ wbase,
                                             int ldb, int n0, int k0, int lane) {
    const int row  = n0 + (lane & 15);
    const int koff = (lane >> 4) * 8;
    const bf16_t* p = wbase + (size_t)row * ldb + k0 + koff;
    union { v16bf v; struct { uint4 lo; uint4 hi; } q; } u;
    u.q.lo = *reinterpret_cast<const uint4*>(p);
    u.q.hi = *reinterpret_cast<const uint4*>(p + 16);
    return u.v;
}

__device__ __forceinline__ v8f wmma_bf16(v16bf a, v16bf b, v8f c) {
    return __builtin_amdgcn_wmma_f32_16x16x32_bf16(false, a, false, b,
                                                   (short)0, c, false, false);
}

__device__ __forceinline__ float fast_sigmoid(float x) {
    return 1.0f / (1.0f + __expf(-x));
}

// ------------------------------ pack kernels -------------------------------
__global__ __launch_bounds__(256) void pack_x_kernel(
    const float* __restrict__ x, const float* __restrict__ tm,
    bf16_t* __restrict__ xbf, int total) {
    int i = blockIdx.x * blockDim.x + threadIdx.x;
    if (i >= total) return;
    int k  = i % KP;
    int tb = i / KP;
    float v = 0.0f;
    if (k < Dn)       v = x[(size_t)tb * Dn + k];
    else if (k == Dn) v = tm[tb];
    xbf[i] = (bf16_t)v;
}

__global__ __launch_bounds__(256) void pack_wih_kernel(
    const float* __restrict__ w, bf16_t* __restrict__ wbf, int total) {
    int i = blockIdx.x * blockDim.x + threadIdx.x;
    if (i >= total) return;
    int k = i % KP;
    int r = i / KP;
    float v = (k < DI) ? w[(size_t)r * DI + k] : 0.0f;
    wbf[i] = (bf16_t)v;
}

__global__ __launch_bounds__(256) void pack_whh_kernel(
    const float* __restrict__ w, bf16_t* __restrict__ wbf, int total) {
    int i = blockIdx.x * blockDim.x + threadIdx.x;
    if (i >= total) return;
    wbf[i] = (bf16_t)w[i];
}

__global__ __launch_bounds__(256) void pack_bias_kernel(
    const float* __restrict__ bih, const float* __restrict__ bhh,
    float* __restrict__ bias, int total) {
    int i = blockIdx.x * blockDim.x + threadIdx.x;
    if (i >= total) return;
    bias[i] = bih[i] + bhh[i];
}

__global__ __launch_bounds__(256) void init_state_kernel(
    bf16_t* __restrict__ hbf, float* __restrict__ c,
    unsigned int* __restrict__ barrier_cnt, int total) {
    int i = blockIdx.x * blockDim.x + threadIdx.x;
    if (i >= total) return;
    hbf[i] = (bf16_t)0.0f;
    c[i]   = 0.0f;
    if (i == 0) *barrier_cnt = 0u;
}

// ------------------------- persistent LSTM kernel --------------------------
__global__ __launch_bounds__(BLKTHREADS) void lstm_persistent_kernel(
    const bf16_t* __restrict__ xbf,     // [T*B, KP]
    const bf16_t* __restrict__ wih,     // [G, KP]
    const bf16_t* __restrict__ whh,     // [G, H]
    const float*  __restrict__ bias,    // [G]
    bf16_t* __restrict__ hbuf0,         // [B, H] bf16 ping
    bf16_t* __restrict__ hbuf1,         // [B, H] bf16 pong
    float*  __restrict__ c_st,          // [B, H] f32 (lane-owned)
    float*  __restrict__ out,           // [T, B, H] f32
    float*  __restrict__ h_final,       // [B, H]
    float*  __restrict__ c_final,       // [B, H]
    unsigned int* __restrict__ barrier_cnt) {
    // W_hh slab, WMMA-fragment layout: slot(gate,kiter) -> 32 lanes x 32B.
    __shared__ __align__(16) unsigned char ldsw[64 * 1024];
#if USE_ASYNC_X
    // Double-buffered x_t tile: [B][KPL] bf16, filled by async global->LDS.
    __shared__ __align__(16) unsigned char ldsx[2][Bn * KPL * 2];
#endif

    const int tid   = threadIdx.x;
    const int lane  = tid & 31;
    const int wl    = tid >> 5;         // 0..3 -> m-tile
    const int jtile = blockIdx.x;       // 0..31
    const int m0  = wl * 16;
    const int j0  = jtile * 16;
    const int nlo = lane & 15;

#if USE_ASYNC_X
    // Issue one step's x tile: 64 rows x 36 chunks of 16B = 2304 chunks;
    // 128 threads x 18 chunks -> exactly 18 async instrs per wave.
    auto async_copy_x = [&](const bf16_t* __restrict__ src,
                            unsigned char* dstbase) {
#pragma unroll
        for (int i = 0; i < 18; ++i) {
            const int c   = tid + i * BLKTHREADS;   // 0..2303
            const int row = c / 36;
            const int col = c % 36;                 // 16-B chunk in row
            const bf16_t* g = src + row * KP + col * 8;
            unsigned char* l = dstbase + row * (KPL * 2) + col * 16;
            __builtin_amdgcn_global_load_async_to_lds_b128(
                (gas_v4i_p)(void*)g, (las_v4i_p)(void*)l, 0, 0);
        }
    };
#endif

    // ---- prologue: W_hh -> LDS (once per run), bias -> registers ----
    for (int slot = wl; slot < 64; slot += NWAVE) {
        const int gate  = slot >> 4;
        const int kiter = slot & 15;
        union { v16bf v; struct { uint4 lo; uint4 hi; } q; } u;
        u.v = load_b_frag(whh, Hn, gate * Hn + j0, kiter * 32, lane);
        uint4* dst = reinterpret_cast<uint4*>(ldsw + slot * 1024 + lane * 32);
        dst[0] = u.q.lo;
        dst[1] = u.q.hi;
    }
#if USE_ASYNC_X
    async_copy_x(xbf + (size_t)(Tn - 1) * Bn * KP, ldsx[0]);   // s = 0 buffer
#endif
    float bv[4];
#pragma unroll
    for (int g = 0; g < 4; ++g) bv[g] = bias[g * Hn + j0 + nlo];
    __syncthreads();

    const int mrow = m0 + (lane >> 4) * 8;
    const int ncol = j0 + nlo;

#pragma unroll 1
    for (int s = 0; s < Tn; ++s) {
        const int t = Tn - 1 - s;
        const bf16_t* h_in  = (s & 1) ? hbuf1 : hbuf0;
        bf16_t*       h_out = (s & 1) ? hbuf0 : hbuf1;
        const int t_next = (t > 0) ? (t - 1) : 0;

#if USE_ASYNC_X
        // Kick next step's tile into the other buffer, then make the current
        // one resident: after issuing 18 new asyncs, waiting for <=18 means
        // the previous 18 (current buffer, in-order) have completed.
        async_copy_x(xbf + (size_t)t_next * Bn * KP, ldsx[(s + 1) & 1]);
        __builtin_amdgcn_s_wait_asynccnt(18);
        __syncthreads();                    // all 4 waves' chunks visible
        const unsigned char* xl = ldsx[s & 1];
#else
        const bf16_t* x_t = xbf + (size_t)t * Bn * KP;
#endif

        v8f acc[4];
#pragma unroll
        for (int g = 0; g < 4; ++g)
            acc[g] = (v8f){bv[g], bv[g], bv[g], bv[g], bv[g], bv[g], bv[g], bv[g]};

        // ---- segment 1: x_t @ W_ih^T  (K = 288) ----
#pragma unroll
        for (int k0 = 0; k0 < KP; k0 += 32) {
#if USE_ASYNC_X
            const v16bf a = load_a_frag_lds(xl, m0, k0, lane);
#else
            const v16bf a = load_a_frag(x_t, KP, m0, k0, lane);
#endif
            v16bf bb[4];
#pragma unroll
            for (int g = 0; g < 4; ++g)
                bb[g] = load_b_frag(wih, KP, g * Hn + j0, k0, lane);
#pragma unroll
            for (int g = 0; g < 4; ++g)
                acc[g] = wmma_bf16(a, bb[g], acc[g]);
        }

        // ---- segment 2: h @ W_hh^T  (K = 512, B frags from LDS) ----
#pragma unroll
        for (int kiter = 0; kiter < 16; ++kiter) {
            const v16bf a = load_a_frag(h_in, Hn, m0, kiter * 32, lane);
#pragma unroll
            for (int g = 0; g < 4; ++g) {
                const uint4* src = reinterpret_cast<const uint4*>(
                    ldsw + (g * 16 + kiter) * 1024 + lane * 32);
                union { v16bf v; struct { uint4 lo; uint4 hi; } q; } u;
                u.q.lo = src[0];
                u.q.hi = src[1];
                acc[g] = wmma_bf16(a, u.v, acc[g]);
            }
        }

        // ---- epilogue: gate nonlinearity + cell update (lane-local) ----
        float* out_t = out + (size_t)t * Bn * Hn;
        const int is_last = (t == 0);
#pragma unroll
        for (int r = 0; r < 8; ++r) {
            const int idx = (mrow + r) * Hn + ncol;
            const float iv = fast_sigmoid(acc[0][r]);
            const float fv = fast_sigmoid(acc[1][r]);
            const float gv = tanhf(acc[2][r]);
            const float ov = fast_sigmoid(acc[3][r]);
            const float cold = c_st[idx];
            const float cnew = fv * cold + iv * gv;
            const float hv   = ov * tanhf(cnew);
            c_st[idx]  = cnew;
            out_t[idx] = hv;
            h_out[idx] = (bf16_t)hv;
            if (is_last) {
                h_final[idx] = hv;
                c_final[idx] = cnew;
            }
        }

#if !USE_ASYNC_X
        // Software prefetch of next step's x tile (global_prefetch_b8) so the
        // lines are L2/near-cache resident when the barrier opens.
        {
            const char* nx = (const char*)(xbf + (size_t)t_next * Bn * KP);
            const int pbase = tid * 288;            // 36864 B / 128 threads
            __builtin_prefetch(nx + pbase, 0, 1);
            __builtin_prefetch(nx + pbase + 128, 0, 1);
            __builtin_prefetch(nx + pbase + 256, 0, 1);
        }
#endif

        // ---- device-wide step barrier (monotonic counter, no reset) ----
        __threadfence();                    // release h_out writes
        __syncthreads();
        if (tid == 0) {
            __hip_atomic_fetch_add(barrier_cnt, 1u, __ATOMIC_RELEASE,
                                   __HIP_MEMORY_SCOPE_AGENT);
            const unsigned int target = (unsigned int)(s + 1) * (unsigned int)NBLK;
            while (__hip_atomic_load(barrier_cnt, __ATOMIC_ACQUIRE,
                                     __HIP_MEMORY_SCOPE_AGENT) < target) {
                __builtin_amdgcn_s_sleep(2);
            }
        }
        __syncthreads();
        __threadfence();                    // acquire: drop stale near-cache
    }
}

// ------------------------------- launcher ----------------------------------
extern "C" void kernel_launch(void* const* d_in, const int* in_sizes, int n_in,
                              void* d_out, int out_size, void* d_ws, size_t ws_size,
                              hipStream_t stream) {
    const float* x    = (const float*)d_in[0];   // [T,B,D]
    const float* tm   = (const float*)d_in[1];   // [T,B,1]
    const float* W_ih = (const float*)d_in[2];   // [4H, D+1]
    const float* W_hh = (const float*)d_in[3];   // [4H, H]
    const float* b_ih = (const float*)d_in[4];   // [4H]
    const float* b_hh = (const float*)d_in[5];   // [4H]
    float* out = (float*)d_out;                  // outputs[T,B,H] ++ h[B,H] ++ c[B,H]

    char* ws = (char*)d_ws;
    size_t off = 0;
    auto carve = [&](size_t bytes) {
        void* p = ws + off;
        off += (bytes + 255) & ~(size_t)255;
        return p;
    };
    bf16_t* xbf   = (bf16_t*)carve((size_t)Tn * Bn * KP * sizeof(bf16_t));
    bf16_t* wihbf = (bf16_t*)carve((size_t)Gn * KP * sizeof(bf16_t));
    bf16_t* whhbf = (bf16_t*)carve((size_t)Gn * Hn * sizeof(bf16_t));
    float*  bias  = (float*) carve((size_t)Gn * sizeof(float));
    bf16_t* hbf0  = (bf16_t*)carve((size_t)Bn * Hn * sizeof(bf16_t));
    bf16_t* hbf1  = (bf16_t*)carve((size_t)Bn * Hn * sizeof(bf16_t));
    float*  c_st  = (float*) carve((size_t)Bn * Hn * sizeof(float));
    unsigned int* cnt = (unsigned int*)carve(sizeof(unsigned int));
    (void)ws_size; (void)in_sizes; (void)n_in; (void)out_size;

    // ---- pack / init (recomputed every call: deterministic, no caching) ----
    {
        int total = Tn * Bn * KP;
        pack_x_kernel<<<(total + 255) / 256, 256, 0, stream>>>(x, tm, xbf, total);
    }
    {
        int total = Gn * KP;
        pack_wih_kernel<<<(total + 255) / 256, 256, 0, stream>>>(W_ih, wihbf, total);
    }
    {
        int total = Gn * Hn;
        pack_whh_kernel<<<(total + 255) / 256, 256, 0, stream>>>(W_hh, whhbf, total);
    }
    {
        int total = Gn;
        pack_bias_kernel<<<(total + 255) / 256, 256, 0, stream>>>(b_ih, b_hh, bias, total);
    }
    {
        int total = Bn * Hn;
        init_state_kernel<<<(total + 255) / 256, 256, 0, stream>>>(hbf0, c_st, cnt, total);
    }

    // ---- one persistent launch runs the whole reverse scan ----
    float* h_final = out + (size_t)Tn * Bn * Hn;
    float* c_final = h_final + (size_t)Bn * Hn;
    lstm_persistent_kernel<<<NBLK, BLKTHREADS, 0, stream>>>(
        xbf, wihbf, whhbf, bias, hbf0, hbf1, c_st, out, h_final, c_final, cnt);
}
